// SO3_Linear_28183575396481
// MI455X (gfx1250) — matrix operand order repaired and checked by reading, not compiled, not run on hardware
//
#include <hip/hip_runtime.h>
#include <hip/hip_bf16.h>

// ---------------------------------------------------------------------------
// SO3 Linear: y[b,o,m] = sum_i W_l[o,i] * x[b,i,m]  for l=0..3, d=2l+1
// bf16x3 split-precision WMMA (hi*hi + hi*lo + lo*hi, f32 accumulate).
// Double-buffered LDS staging + software-pipelined global gather.
// ---------------------------------------------------------------------------

typedef __attribute__((ext_vector_type(16))) __bf16        v16bf;
typedef __attribute__((ext_vector_type(8)))  float         v8f;
typedef __attribute__((ext_vector_type(8)))  unsigned short v8us;
typedef __attribute__((ext_vector_type(16))) unsigned short v16us;

#define MUL   256
#define BATCH 20000

__device__ __forceinline__ unsigned f2u(float f) { return __float_as_uint(f); }
__device__ __forceinline__ float    u2f(unsigned u) { return __uint_as_float(u); }

// Split f32 into bf16 hi (truncation, exact top 16 bits) and bf16 lo
// (truncation of the exact residual). hi+lo keeps >= 16 significant bits (rel. err < 2^-15).
__device__ __forceinline__ void bf16_split(float v, unsigned short& hi, unsigned short& lo) {
    unsigned u = f2u(v);
    hi = (unsigned short)(u >> 16);
    float res = v - u2f(u & 0xFFFF0000u);   // exact in f32
    lo = (unsigned short)(f2u(res) >> 16);
}

// ---------------------------------------------------------------------------
// Pre-kernel: split all four W (256x256 f32, row-major [o][i]) into bf16
// hi/lo planes in workspace.  Layout: Whi[l*65536 + o*256 + k], same for Wlo.
// ---------------------------------------------------------------------------
__global__ __launch_bounds__(256) void wsplit_kernel(
    const float* __restrict__ W0, const float* __restrict__ W1,
    const float* __restrict__ W2, const float* __restrict__ W3,
    unsigned short* __restrict__ Whi, unsigned short* __restrict__ Wlo)
{
    int e = blockIdx.x * 256 + threadIdx.x;     // 0 .. 4*65536-1
    const float* Ws[4] = {W0, W1, W2, W3};
    float w = Ws[e >> 16][e & 65535];
    unsigned short hi, lo;
    bf16_split(w, hi, lo);
    Whi[e] = hi;
    Wlo[e] = lo;
}

// ---------------------------------------------------------------------------
// Main kernel, one instantiation per l.  WG = 256 threads = 8 wave32.
// WG tile: 64 rows ((b,m) pairs) x 256 output channels, K = 256 in 8
// chunks of 32.  Wave w: row-tile = w&3 (16 rows), col-half = w>>2 (128 o).
// LDS A-tile is double buffered; next chunk's global gather is issued
// before the current chunk's WMMA block so HBM latency hides under MMA.
// ---------------------------------------------------------------------------
template<int L>
__global__ __launch_bounds__(256) void so3_wmma_kernel(
    const float* __restrict__ x,
    const unsigned short* __restrict__ Whi,   // [o*256 + k] bf16 bits, this l
    const unsigned short* __restrict__ Wlo,
    const float* __restrict__ bias,
    float* __restrict__ out)
{
    constexpr int  D    = 2 * L + 1;
    constexpr int  OFF  = (L == 0) ? 0 : (L == 1) ? 256 : (L == 2) ? 1024 : 2304;
    constexpr long ROWS = (long)BATCH * D;
    constexpr int  LDA  = 40;                 // 32 K-halfs + 8 pad (16B-aligned rows)
    constexpr int  BUF  = 64 * LDA;           // halfs per buffer

    __shared__ __align__(16) unsigned short Ahi[2 * BUF];
    __shared__ __align__(16) unsigned short Alo[2 * BUF];

    const int tid     = threadIdx.x;
    const int lane    = tid & 31;
    const int wave    = tid >> 5;
    const int rowTile = wave & 3;             // 0..3 -> 16-row slab
    const int colHalf = wave >> 2;            // 0..1 -> 128-col half

    // ---- fill-role mapping: thread t stages row t>>2, k-subrange (t&3)*8 ----
    const int  frow   = tid >> 2;             // 0..63
    const int  fk     = (tid & 3) * 8;        // 0,8,16,24 (local k)
    const long fr     = (long)blockIdx.x * 64 + frow;
    const bool fvalid = fr < ROWS;
    const int  fb     = (int)(fr / D);
    const int  fm     = (int)(fr % D);
    const float* px   = x + (size_t)fb * 4096 + OFF + fm;   // + i*D per k

    // ---- compute-role lane mapping (ISA rows/cols, contiguous K halves) ----
    const int aRow  = rowTile * 16 + (lane & 15);           // A: lane&15 = row
    const int abase = (lane < 16) ? 0 : 16;                 // A K-half select
    const int bkoff = (lane < 16) ? 0 : 16;                 // B K-half select
    const int oLane = colHalf * 128 + (lane & 15);          // B/C: lane&15 = col

    v8f acc[8];
    v8f vzero = {0.f, 0.f, 0.f, 0.f, 0.f, 0.f, 0.f, 0.f};
#pragma unroll
    for (int t = 0; t < 8; ++t) acc[t] = vzero;

    // prologue: gather chunk 0
    float v[8];
#pragma unroll
    for (int u = 0; u < 8; ++u)
        v[u] = fvalid ? px[(size_t)(fk + u) * D] : 0.0f;

#pragma unroll 2
    for (int kc = 0; kc < 8; ++kc) {
        const int k0   = kc * 32;
        const int sbuf = (kc & 1) * BUF;

        // split current chunk and stage into LDS buffer kc&1
        v8us hi8, lo8;
#pragma unroll
        for (int u = 0; u < 8; ++u) {
            unsigned short h, l;
            bf16_split(v[u], h, l);
            hi8[u] = h; lo8[u] = l;
        }
        *(v8us*)(&Ahi[sbuf + frow * LDA + fk]) = hi8;
        *(v8us*)(&Alo[sbuf + frow * LDA + fk]) = lo8;
        __syncthreads();

        // issue next chunk's global gather now; its s_wait sinks below the MMAs
        if (kc < 7) {
#pragma unroll
            for (int u = 0; u < 8; ++u)
                v[u] = fvalid ? px[(size_t)(k0 + 32 + fk + u) * D] : 0.0f;
        }

        // A fragments: lane<16 holds K {0..15}, lane>=16 holds {16..31}
        v8us ah0 = *(const v8us*)(&Ahi[sbuf + aRow * LDA + abase]);
        v8us ah1 = *(const v8us*)(&Ahi[sbuf + aRow * LDA + abase + 8]);
        v8us al0 = *(const v8us*)(&Alo[sbuf + aRow * LDA + abase]);
        v8us al1 = *(const v8us*)(&Alo[sbuf + aRow * LDA + abase + 8]);
        v16us ahu = __builtin_shufflevector(ah0, ah1, 0,1,2,3,4,5,6,7,8,9,10,11,12,13,14,15);
        v16us alu = __builtin_shufflevector(al0, al1, 0,1,2,3,4,5,6,7,8,9,10,11,12,13,14,15);
        v16bf Af_hi = __builtin_bit_cast(v16bf, ahu);
        v16bf Af_lo = __builtin_bit_cast(v16bf, alu);

#pragma unroll
        for (int ot = 0; ot < 8; ++ot) {
            const int o = oLane + ot * 16;
            // B fragments: lane = column o, 16 consecutive K per lane (row-major W)
            v16us bhu = *(const v16us*)(Whi + (size_t)o * 256 + k0 + bkoff);
            v16us blu = *(const v16us*)(Wlo + (size_t)o * 256 + k0 + bkoff);
            v16bf Bf_hi = __builtin_bit_cast(v16bf, bhu);
            v16bf Bf_lo = __builtin_bit_cast(v16bf, blu);
            acc[ot] = __builtin_amdgcn_wmma_f32_16x16x32_bf16(
                false, Af_hi, false, Bf_hi, (short)0, acc[ot], false, false);
            acc[ot] = __builtin_amdgcn_wmma_f32_16x16x32_bf16(
                false, Af_hi, false, Bf_lo, (short)0, acc[ot], false, false);
            acc[ot] = __builtin_amdgcn_wmma_f32_16x16x32_bf16(
                false, Af_lo, false, Bf_hi, (short)0, acc[ot], false, false);
        }
    }

    // ---- epilogue: C layout VGPR j = row (lane<16: M=j, else M=8+j), lane&15 = col ----
    const int  rsel  = (lane < 16) ? 0 : 8;
    const long rbase = (long)blockIdx.x * 64 + rowTile * 16 + rsel;
#pragma unroll
    for (int ot = 0; ot < 8; ++ot) {
        const int   o  = oLane + ot * 16;
        const float bv = (L == 0) ? bias[o] : 0.0f;
#pragma unroll
        for (int j = 0; j < 8; ++j) {
            const long r = rbase + j;
            if (r < ROWS) {
                const int b = (int)(r / D);
                const int m = (int)(r % D);
                out[(size_t)b * 4096 + OFF + (size_t)o * D + m] = acc[ot][j] + bv;
            }
        }
    }
}

// ---------------------------------------------------------------------------
extern "C" void kernel_launch(void* const* d_in, const int* in_sizes, int n_in,
                              void* d_out, int out_size, void* d_ws, size_t ws_size,
                              hipStream_t stream) {
    (void)in_sizes; (void)n_in; (void)out_size; (void)ws_size;
    const float* x    = (const float*)d_in[0];
    const float* W0   = (const float*)d_in[1];
    const float* W1   = (const float*)d_in[2];
    const float* W2   = (const float*)d_in[3];
    const float* W3   = (const float*)d_in[4];
    const float* bias = (const float*)d_in[5];
    float* out = (float*)d_out;

    unsigned short* Whi = (unsigned short*)d_ws;          // 4*65536 bf16
    unsigned short* Wlo = Whi + 4 * 65536;                // 4*65536 bf16 (1 MB total)

    wsplit_kernel<<<1024, 256, 0, stream>>>(W0, W1, W2, W3, Whi, Wlo);

    const int b0 = (int)(((long)BATCH * 1 + 63) / 64);    // 313
    const int b1 = (int)(((long)BATCH * 3 + 63) / 64);    // 938
    const int b2 = (int)(((long)BATCH * 5 + 63) / 64);    // 1563
    const int b3 = (int)(((long)BATCH * 7 + 63) / 64);    // 2188

    so3_wmma_kernel<0><<<b0, 256, 0, stream>>>(x, Whi + 0 * 65536, Wlo + 0 * 65536, bias, out);
    so3_wmma_kernel<1><<<b1, 256, 0, stream>>>(x, Whi + 1 * 65536, Wlo + 1 * 65536, bias, out);
    so3_wmma_kernel<2><<<b2, 256, 0, stream>>>(x, Whi + 2 * 65536, Wlo + 2 * 65536, bias, out);
    so3_wmma_kernel<3><<<b3, 256, 0, stream>>>(x, Whi + 3 * 65536, Wlo + 3 * 65536, bias, out);
}
